// SNNConvNet_59090160058401
// MI455X (gfx1250) — matrix operand order repaired and checked by reading, not compiled, not run on hardware
//
#include <hip/hip_runtime.h>
#include <hip/hip_bf16.h>
#include <stdint.h>

typedef __attribute__((ext_vector_type(16))) _Float16 v16h;
typedef __attribute__((ext_vector_type(8)))  float    v8f;

typedef unsigned int tdm_v4u __attribute__((ext_vector_type(4)));
typedef int          tdm_v4i __attribute__((ext_vector_type(4)));
typedef int          tdm_v8i __attribute__((ext_vector_type(8)));

// ---------------------------------------------------------------------------
// TDM helper (clang-23 / therock 6-arg builtin, confirmed by compile probe:
// "expected 6"). Descriptor per CDNA5 ISA: group0: count=1 | lds_addr |
// global_addr | type=2 ; group1: data_size=2B, tensor_dim0=tile_dim0=nhalfs,
// tensor_dim1=tile_dim1=1, dim0_stride=nhalfs. Degenerate 1-row 2D tile ->
// contiguous bulk copy global->LDS tracked on TENSORcnt.
// ---------------------------------------------------------------------------
#if __has_builtin(__builtin_amdgcn_tensor_load_to_lds)
#define HAVE_TDM 1
__device__ __forceinline__ void tdm_load_halfs(const _Float16* gsrc, unsigned lds_off,
                                               int nhalfs) {
    unsigned long long ga = (unsigned long long)(uintptr_t)gsrc;
    tdm_v4u g0 = { 1u,                                   // count=1, user mode
                   lds_off,                              // LDS byte address
                   (unsigned)ga,                         // global_addr[31:0]
                   (unsigned)((ga >> 32) & 0x01ffffffu) | 0x80000000u }; // [56:32]|type=2
    tdm_v8i g1 = { (int)(1u << 16),                                  // data_size=1 (2B)
                   (int)((unsigned)(nhalfs & 0xffff) << 16),         // tensor_dim0 lo16
                   (int)(((unsigned)nhalfs >> 16) | (1u << 16)),     // dim0 hi16, dim1=1
                   (int)((unsigned)(nhalfs & 0xffff) << 16),         // tile_dim0
                   1,                                                // tile_dim1=1
                   nhalfs,                                           // dim0_stride lo32
                   0, 0 };
    tdm_v4i gz4 = { 0, 0, 0, 0 };
    tdm_v8i gz8 = { 0, 0, 0, 0, 0, 0, 0, 0 };
    __builtin_amdgcn_tensor_load_to_lds(g0, g1, gz4, gz4, gz8, 0);
}
#else
#define HAVE_TDM 0
#endif

__device__ __forceinline__ void wait_tensor0() {
#if __has_builtin(__builtin_amdgcn_s_wait_tensorcnt)
    __builtin_amdgcn_s_wait_tensorcnt((short)0);
#else
    asm volatile("s_wait_tensorcnt 0x0" ::: "memory");
#endif
}

// ---------------------------------------------------------------------------
// Kernel 1: conv (2->8ch, 3x3, SAME on B x 2 x 512 x 32) + 4-step conv-LIF
// evaluated analytically (conv input identical per timestep). Writes fp32
// spike record (B,T,8,512,32) and packed u8 GEMM rows [(t*128+b), k].
// ---------------------------------------------------------------------------
__global__ __launch_bounds__(256)
void conv_lif_kernel(const float* __restrict__ x, const float* __restrict__ wconv,
                     float* __restrict__ conv_spk, unsigned char* __restrict__ S_u8) {
    __shared__ float wc[144];
    if (threadIdx.x < 144) wc[threadIdx.x] = wconv[threadIdx.x];
    __syncthreads();

    int tid = blockIdx.x * 256 + threadIdx.x;          // 16,777,216 total
    int w  = tid & 31;
    int h  = (tid >> 5) & 511;
    int oc = (tid >> 14) & 7;
    int b  = tid >> 17;

    float acc = 0.f;
    #pragma unroll
    for (int ic = 0; ic < 2; ++ic) {
        const float* xp = x + (size_t)(b * 2 + ic) * 512 * 32;
        const float* wp = wc + (oc * 2 + ic) * 9;
        #pragma unroll
        for (int kh = 0; kh < 3; ++kh) {
            int hh = h + kh - 1;
            if (hh < 0 || hh > 511) continue;
            #pragma unroll
            for (int kw = 0; kw < 3; ++kw) {
                int ww = w + kw - 1;
                if (ww < 0 || ww > 31) continue;
                acc += xp[hh * 32 + ww] * wp[kh * 3 + kw];
            }
        }
    }

    size_t pix = (size_t)oc * 16384 + (size_t)h * 32 + w;
    float v = 0.f;
    #pragma unroll
    for (int t = 0; t < 4; ++t) {
        v = 0.5f * v + acc;
        float s = (v >= 1.0f) ? 1.f : 0.f;
        v = (v >= 1.0f) ? 0.f : v;
        conv_spk[(size_t)(b * 4 + t) * 131072 + pix] = s;
        S_u8[(size_t)(t * 128 + b) * 131072 + pix] = (unsigned char)s;
    }
}

// ---------------------------------------------------------------------------
// Kernel 2: one-shot weight convert fp32 -> f16 into tiled-transposed layout:
// tile (nt,kt) = 64n x 32k stored [n][k]-contiguous (4 KB), tiles contiguous
// in kt for fixed nt. Coalesced read, identity LDS staging, coalesced 16B writes.
// grid = (K/32, N/64), 256 threads.
// ---------------------------------------------------------------------------
__global__ __launch_bounds__(256)
void convert_wt_kernel(const float* __restrict__ W, _Float16* __restrict__ Wt,
                       int N, int K) {
    __shared__ __align__(32) _Float16 Lt[2048];
    const int kt = blockIdx.x, nt = blockIdx.y, t = threadIdx.x;
    const int nin = t >> 2, kq = (t & 3) * 8;

    const float* wp = W + (size_t)(nt * 64 + nin) * (size_t)K + kt * 32 + kq;
    float4 lo = *(const float4*)wp;
    float4 hi = *(const float4*)(wp + 4);
    Lt[nin * 32 + kq + 0] = (_Float16)lo.x;
    Lt[nin * 32 + kq + 1] = (_Float16)lo.y;
    Lt[nin * 32 + kq + 2] = (_Float16)lo.z;
    Lt[nin * 32 + kq + 3] = (_Float16)lo.w;
    Lt[nin * 32 + kq + 4] = (_Float16)hi.x;
    Lt[nin * 32 + kq + 5] = (_Float16)hi.y;
    Lt[nin * 32 + kq + 6] = (_Float16)hi.z;
    Lt[nin * 32 + kq + 7] = (_Float16)hi.w;
    __syncthreads();

    _Float16* dst = Wt + ((size_t)nt * (size_t)(K / 32) + kt) * 2048 + t * 8;
    *(float4*)dst = *(const float4*)&Lt[t * 8];
}

// ---------------------------------------------------------------------------
// Kernel 3: C[M,N] = A_spikes[M,K](u8) * W^T via pre-tiled f16 weights.
// 8 waves; wave = 16(M) x 64(N) strip. K-step 64: TDM double-buffered 8 KB
// tile stage (wave 0 issues tensor_load_to_lds, s_wait_tensorcnt, 1 barrier
// per step), 8 x v_wmma_f32_16x16x32_f16 per step. B frag = contiguous 32B
// ds read (lane&15 == N-col). Requires M%128==0, N%64==0, K%64==0.
// ---------------------------------------------------------------------------
__global__ __launch_bounds__(256)
void gemm_spk_kernel(const unsigned char* __restrict__ A, const _Float16* __restrict__ Wt,
                     float* __restrict__ C, int M, int N, int K) {
    __shared__ __align__(32) _Float16 Lt[2][4096];   // 2 x 8KB double buffer

    const int tid  = threadIdx.x;
    const int lane = tid & 31;
    const int wave = tid >> 5;
    const int n0   = blockIdx.x * 64;
    const int m0w  = blockIdx.y * 128 + wave * 16;
    const int nk   = K / 64;

    const unsigned char* Arow = A + (size_t)(m0w + (lane & 15)) * (size_t)K;
    const int ahalf = (lane >> 4) * 16;
    const _Float16* Wstrip = Wt + (size_t)(n0 >> 6) * (size_t)(K / 32) * 2048;

    v8f c0 = {}, c1 = {}, c2 = {}, c3 = {};

#if HAVE_TDM
    if (wave == 0) tdm_load_halfs(Wstrip, (unsigned)(uintptr_t)&Lt[0][0], 4096);
#endif

    for (int j = 0; j < nk; ++j) {
#if HAVE_TDM
        const int p = j & 1;
        if (wave == 0) wait_tensor0();
        __syncthreads();                               // tile p ready for all waves
        if (wave == 0 && (j + 1) < nk)
            tdm_load_halfs(Wstrip + (size_t)(j + 1) * 4096,
                           (unsigned)(uintptr_t)&Lt[p ^ 1][0], 4096);
#else
        const int p = 0;
        __syncthreads();                               // previous reads done
        {   // cooperative 8KB copy of pre-tiled f16 tile
            const float4* src = (const float4*)(Wstrip + (size_t)j * 4096);
            float4* dst = (float4*)&Lt[0][0];
            dst[tid] = src[tid];
            dst[tid + 256] = src[tid + 256];
        }
        __syncthreads();
#endif
        __builtin_prefetch(Arow + (size_t)j * 64 + 64, 0, 0);  // next spike chunk

        const _Float16* Lb = &Lt[p][0];
        #pragma unroll
        for (int q = 0; q < 2; ++q) {
            unsigned long long lo = *(const unsigned long long*)(Arow + j * 64 + q * 32 + ahalf);
            unsigned long long hi = *(const unsigned long long*)(Arow + j * 64 + q * 32 + 8 + ahalf);
            union { v16h h; unsigned short u[16]; } a;
            #pragma unroll
            for (int i = 0; i < 8; ++i) {
                a.u[i]     = (unsigned short)((lo >> (8 * i)) & 1ull) * (unsigned short)0x3C00;
                a.u[8 + i] = (unsigned short)((hi >> (8 * i)) & 1ull) * (unsigned short)0x3C00;
            }
            const v16h* bp = (const v16h*)(Lb + q * 2048 + (lane & 15) * 32 + (lane >> 4) * 16);
            c0 = __builtin_amdgcn_wmma_f32_16x16x32_f16(false, a.h, false, bp[0], (short)0, c0, false, false);
            c1 = __builtin_amdgcn_wmma_f32_16x16x32_f16(false, a.h, false, bp[32], (short)0, c1, false, false);
            c2 = __builtin_amdgcn_wmma_f32_16x16x32_f16(false, a.h, false, bp[64], (short)0, c2, false, false);
            c3 = __builtin_amdgcn_wmma_f32_16x16x32_f16(false, a.h, false, bp[96], (short)0, c3, false, false);
        }
    }

    // C/D layout: VGPR r -> M = r + 8*(lane>>4), N = lane&15
    const int rbase = m0w + ((lane >> 4) << 3);
    const int cbase = n0 + (lane & 15);
    #pragma unroll
    for (int r = 0; r < 8; ++r) {
        size_t row = (size_t)(rbase + r) * (size_t)N;
        C[row + cbase]      = c0[r];
        C[row + cbase + 16] = c1[r];
        C[row + cbase + 32] = c2[r];
        C[row + cbase + 48] = c3[r];
    }
}

// ---------------------------------------------------------------------------
// Kernel 4/6: sequential LIF over T=4 on (128,512) states.
// ---------------------------------------------------------------------------
__global__ __launch_bounds__(256)
void lif_fc_kernel(const float* __restrict__ z, float* __restrict__ spk_rec,
                   unsigned char* __restrict__ s_u8, float* __restrict__ mean_out) {
    int tid = blockIdx.x * 256 + threadIdx.x;   // 65536
    int n = tid & 511;
    int b = tid >> 9;
    float v = 0.f, acc = 0.f;
    #pragma unroll
    for (int t = 0; t < 4; ++t) {
        float zz = z[(size_t)(t * 128 + b) * 512 + n];
        v = 0.5f * v + zz;
        float s = (v >= 1.0f) ? 1.f : 0.f;
        v = (v >= 1.0f) ? 0.f : v;
        spk_rec[(size_t)(b * 4 + t) * 512 + n] = s;
        s_u8[(size_t)(t * 128 + b) * 512 + n] = (unsigned char)s;
        acc += s;
    }
    if (mean_out) mean_out[(size_t)b * 512 + n] = acc * 0.25f;
}

// ---------------------------------------------------------------------------
// Kernel 7: out = mean(s2) @ w_out^T -> (128,16).
// ---------------------------------------------------------------------------
__global__ __launch_bounds__(256)
void out_gemm_kernel(const float* __restrict__ s2m, const float* __restrict__ wout,
                     float* __restrict__ out) {
    int tid = blockIdx.x * 256 + threadIdx.x;   // 2048
    int o = tid & 15;
    int b = tid >> 4;
    const float* sp = s2m + (size_t)b * 512;
    const float* wp = wout + (size_t)o * 512;
    float acc = 0.f;
    #pragma unroll 4
    for (int k = 0; k < 512; ++k) acc += sp[k] * wp[k];
    out[tid] = acc;
}

// ---------------------------------------------------------------------------
extern "C" void kernel_launch(void* const* d_in, const int* in_sizes, int n_in,
                              void* d_out, int out_size, void* d_ws, size_t ws_size,
                              hipStream_t stream) {
    const float* x      = (const float*)d_in[0];  // (128,2,64,8,32)
    const float* w_conv = (const float*)d_in[1];  // (8,2,3,3)
    const float* w1     = (const float*)d_in[2];  // (512,131072)
    const float* w2     = (const float*)d_in[3];  // (512,512)
    const float* w_out  = (const float*)d_in[4];  // (16,512)

    float* out      = (float*)d_out;              // (128,16)
    float* conv_spk = out + 2048;                 // (128,4,8,512,32)
    float* in_spk   = conv_spk + 67108864;        // (128,4,512)
    float* hid_spk  = in_spk + 262144;            // (128,4,512)

    char* ws = (char*)d_ws;
    unsigned char* S_u8  = (unsigned char*)ws;                  // 67,108,864 B
    _Float16*      w1t   = (_Float16*)(ws + 67108864);          // 134,217,728 B
    _Float16*      w2t   = (_Float16*)(ws + 201326592);         // 524,288 B
    unsigned char* s1_u8 = (unsigned char*)(ws + 201850880);    // 262,144 B
    float* z1  = (float*)(ws + 202113024);                      // 1 MB
    float* z2  = (float*)(ws + 203161600);                      // 1 MB
    float* s2m = (float*)(ws + 204210176);                      // 256 KB

    // 1) conv + conv-LIF spikes for all 4 timesteps
    conv_lif_kernel<<<65536, 256, 0, stream>>>(x, w_conv, conv_spk, S_u8);

    // 2) weight pre-convert to tiled-transposed f16
    convert_wt_kernel<<<dim3(4096, 8), 256, 0, stream>>>(w1, w1t, 512, 131072);
    convert_wt_kernel<<<dim3(16, 8),   256, 0, stream>>>(w2, w2t, 512, 512);

    // 3) z1[T*B,512] = spikes[T*B,131072] @ w1^T  (batched over T, TDM+WMMA)
    gemm_spk_kernel<<<dim3(8, 4), 256, 0, stream>>>(S_u8, w1t, z1, 512, 512, 131072);

    // 4) LIF layer 1
    lif_fc_kernel<<<256, 256, 0, stream>>>(z1, in_spk, s1_u8, nullptr);

    // 5) z2[T*B,512] = s1 @ w2^T
    gemm_spk_kernel<<<dim3(8, 4), 256, 0, stream>>>(s1_u8, w2t, z2, 512, 512, 512);

    // 6) LIF layer 2 + temporal mean
    lif_fc_kernel<<<256, 256, 0, stream>>>(z2, hid_spk, s1_u8, s2m);

    // 7) output projection
    out_gemm_kernel<<<8, 256, 0, stream>>>(s2m, w_out, out);
}
